// ContraAtt_32985348833923
// MI455X (gfx1250) — matrix operand order repaired and checked by reading, not compiled, not run on hardware
//
#include <hip/hip_runtime.h>

#define SDIM 196
#define BDIM 64
#define NDIM 512
#define DDIM 1024
#define HDIM 8
#define MROWS (SDIM*BDIM)     // 12544 rows, divisible by 16
#define INV_SCALE 0.03125f    // 1/sqrt(1024)

typedef __attribute__((ext_vector_type(16))) __bf16          v16bf;
typedef __attribute__((ext_vector_type(16))) unsigned short  v16u;
typedef __attribute__((ext_vector_type(8)))  float           v8f;

__device__ __forceinline__ unsigned short f2bf(float f) {
  unsigned int u = __float_as_uint(f);
  u += 0x7FFFu + ((u >> 16) & 1u);          // round-to-nearest-even
  return (unsigned short)(u >> 16);
}

// ---------------- K1: g = mean_S(input); also common[b][0] = g ----------------
__global__ void k_mean(const float* __restrict__ in, float* __restrict__ g,
                       float* __restrict__ common) {
  int idx = blockIdx.x * 256 + threadIdx.x;        // B*D threads
  int b = idx >> 10, d = idx & 1023;
  float s = 0.f;
  for (int t = 0; t < SDIM; ++t) s += in[((size_t)t*BDIM + b)*DDIM + d];
  s *= (1.0f / SDIM);
  g[idx] = s;
  common[((size_t)b*9)*DDIM + d] = s;
}

// ---------------- K2: Q[b,h,e] = g[b]·Wq[h,:,e] + bq[h,e] ----------------
__global__ void k_qproj(const float* __restrict__ g, const float* __restrict__ Wq,
                        const float* __restrict__ bq, float* __restrict__ Q) {
  int h = blockIdx.z;
  int e = blockIdx.x*64 + (threadIdx.x & 63);
  int b = blockIdx.y*4  + (threadIdx.x >> 6);
  const float* w  = Wq + (size_t)h*DDIM*DDIM;
  const float* gv = g + (size_t)b*DDIM;
  float acc = bq[h*DDIM + e];
  for (int d = 0; d < DDIM; ++d) acc += gv[d] * w[(size_t)d*DDIM + e];
  Q[((size_t)b*HDIM + h)*DDIM + e] = acc;
}

// ---------------- K3: Qk[b,h,d] = Q[b,h]·Wk[h,d,:] ----------------
__global__ void k_qk(const float* __restrict__ Q, const float* __restrict__ Wk,
                     float* __restrict__ Qk) {
  int h = blockIdx.y;
  int d = blockIdx.x*4 + (threadIdx.x >> 6);
  int b = threadIdx.x & 63;
  const float* w = Wk + ((size_t)h*DDIM + d)*DDIM;
  const float* q = Q  + ((size_t)b*HDIM + h)*DDIM;
  float acc = 0.f;
  for (int e = 0; e < DDIM; ++e) acc += q[e] * w[e];
  Qk[((size_t)b*HDIM + h)*DDIM + d] = acc;
}

// ---------------- K3b: qc[b,h] = Q[b,h]·bk[h] ----------------
__global__ void k_qc(const float* __restrict__ Q, const float* __restrict__ bk,
                     float* __restrict__ qc) {
  int idx = blockIdx.x*256 + threadIdx.x;          // 512 = B*H
  int b = idx >> 3, h = idx & 7;
  const float* q  = Q  + ((size_t)b*HDIM + h)*DDIM;
  const float* bb = bk + (size_t)h*DDIM;
  float acc = 0.f;
  for (int e = 0; e < DDIM; ++e) acc += q[e]*bb[e];
  qc[idx] = acc;
}

// ---------------- K4: M[b,h,n] = (gnf[b,n]·Qk[b,h] + qc[b,h]) / 32 ----------------
__global__ void k_logits(const float* __restrict__ gnf, const float* __restrict__ Qk,
                         const float* __restrict__ qc, float* __restrict__ M) {
  __shared__ float sG[32][65];
  __shared__ float sQ[8][64];
  int b = blockIdx.y, n0 = blockIdx.x*32;
  int h = threadIdx.x >> 5, n = threadIdx.x & 31;
  float acc = 0.f;
  for (int dc = 0; dc < DDIM; dc += 64) {
    for (int t = threadIdx.x; t < 32*64; t += 256) {
      int i = t >> 6, j = t & 63;
      sG[i][j] = gnf[((size_t)b*NDIM + n0 + i)*DDIM + dc + j];
    }
    for (int t = threadIdx.x; t < 8*64; t += 256) {
      int hh = t >> 6, j = t & 63;
      sQ[hh][j] = Qk[((size_t)b*HDIM + hh)*DDIM + dc + j];
    }
    __syncthreads();
    #pragma unroll
    for (int j = 0; j < 64; ++j) acc += sG[n][j]*sQ[h][j];
    __syncthreads();
  }
  M[((size_t)b*HDIM + h)*NDIM + n0 + n] = (acc + qc[b*HDIM + h]) * INV_SCALE;
}

// ---------------- K5: softmax over N=512 (in place) ----------------
__global__ void k_softmax(float* __restrict__ M) {
  __shared__ float red[256];
  float* p = M + (size_t)blockIdx.x*NDIM;
  float v0 = p[threadIdx.x], v1 = p[threadIdx.x + 256];
  red[threadIdx.x] = fmaxf(v0, v1); __syncthreads();
  for (int o = 128; o > 0; o >>= 1) {
    if (threadIdx.x < o) red[threadIdx.x] = fmaxf(red[threadIdx.x], red[threadIdx.x+o]);
    __syncthreads();
  }
  float m = red[0]; __syncthreads();
  float e0 = __expf(v0 - m), e1 = __expf(v1 - m);
  red[threadIdx.x] = e0 + e1; __syncthreads();
  for (int o = 128; o > 0; o >>= 1) {
    if (threadIdx.x < o) red[threadIdx.x] += red[threadIdx.x+o];
    __syncthreads();
  }
  float inv = 1.0f / red[0];
  p[threadIdx.x] = e0*inv; p[threadIdx.x+256] = e1*inv;
}

// ---------------- K6: common[b][1+h] = sum_n attn[b,h,n] * gnf[b,n,:] ----------------
__global__ void k_closest(const float* __restrict__ gnf, const float* __restrict__ attn,
                          float* __restrict__ common) {
  __shared__ float sG[64][33];
  __shared__ float sA[8][64];
  int b = blockIdx.y, d0 = blockIdx.x*32;
  int h = threadIdx.x >> 5, dl = threadIdx.x & 31;
  float acc = 0.f;
  for (int nc = 0; nc < NDIM; nc += 64) {
    for (int t = threadIdx.x; t < 64*32; t += 256) {
      int i = t >> 5, j = t & 31;
      sG[i][j] = gnf[((size_t)b*NDIM + nc + i)*DDIM + d0 + j];
    }
    for (int t = threadIdx.x; t < 8*64; t += 256) {
      int hh = t >> 6, j = t & 63;
      sA[hh][j] = attn[((size_t)b*HDIM + hh)*NDIM + nc + j];
    }
    __syncthreads();
    #pragma unroll
    for (int j = 0; j < 64; ++j) acc += sA[h][j]*sG[j][dl];
    __syncthreads();
  }
  common[((size_t)b*9 + 1 + h)*DDIM + d0 + dl] = acc;
}

// ---------------- K7a/b: Y[r,e] = common[r]·W[:,e] + bias[e], r in [0,576) ----------------
__global__ void k_proj576(const float* __restrict__ X, const float* __restrict__ W,
                          const float* __restrict__ bias, float* __restrict__ Y) {
  int e = blockIdx.x*64 + (threadIdx.x & 63);
  int r = blockIdx.y*4  + (threadIdx.x >> 6);
  const float* x = X + (size_t)r*DDIM;
  float acc = bias[e];
  for (int d = 0; d < DDIM; ++d) acc += x[d]*W[(size_t)d*DDIM + e];
  Y[(size_t)r*DDIM + e] = acc;
}

// ---------------- K7c: 9x9 diff attention per b; diff = g - common_info ----------------
__global__ void k_diffatt(const float* __restrict__ Qd, const float* __restrict__ Kd,
                          const float* __restrict__ common, const float* __restrict__ g,
                          float* __restrict__ diffb) {
  __shared__ float sMd[9][12];
  __shared__ float sW[9];
  int b = blockIdx.x;
  int wave = threadIdx.x >> 5, lane = threadIdx.x & 31;
  for (int p = wave; p < 81; p += 8) {
    int m = p / 9, n = p % 9;
    const float* qm = Qd + ((size_t)b*9 + m)*DDIM;
    const float* kn = Kd + ((size_t)b*9 + n)*DDIM;
    float v = 0.f;
    for (int e = lane; e < DDIM; e += 32) v += qm[e]*kn[e];
    for (int o = 16; o > 0; o >>= 1) v += __shfl_xor(v, o, 32);
    if (lane == 0) sMd[m][n] = v * INV_SCALE;
  }
  __syncthreads();
  if (threadIdx.x < 9) {                           // softmax rows
    int m = threadIdx.x;
    float mx = sMd[m][0];
    for (int n = 1; n < 9; ++n) mx = fmaxf(mx, sMd[m][n]);
    float ev[9]; float s = 0.f;
    for (int n = 0; n < 9; ++n) { ev[n] = __expf(sMd[m][n]-mx); s += ev[n]; }
    float inv = 1.f/s;
    for (int n = 0; n < 9; ++n) sMd[m][n] = ev[n]*inv;
  }
  __syncthreads();
  if (threadIdx.x < 9) {                           // w[n] = mean_m attd[m,n]
    int n = threadIdx.x;
    float s = 0.f;
    for (int m = 0; m < 9; ++m) s += sMd[m][n];
    sW[n] = s * (1.0f/9.0f);
  }
  __syncthreads();
  for (int d = threadIdx.x; d < DDIM; d += 256) {
    const float* c = common + (size_t)b*9*DDIM + d;
    float ci = 0.f;
    #pragma unroll
    for (int n = 0; n < 9; ++n) ci += sW[n]*c[(size_t)n*DDIM];
    diffb[(size_t)b*DDIM + d] = g[(size_t)b*DDIM + d] - ci;
  }
}

// ---------------- K8: t[b,n] = upd_b[n] + diff[b]·W2[:,n] ----------------
__global__ void k_tcalc(const float* __restrict__ diffb, const float* __restrict__ updW,
                        const float* __restrict__ updb, float* __restrict__ tb) {
  int b = blockIdx.y;
  int n = blockIdx.x*256 + threadIdx.x;
  const float* w2 = updW + (size_t)DDIM*DDIM + n;
  const float* dv = diffb + (size_t)b*DDIM;
  float acc = updb[n];
  for (int k = 0; k < DDIM; ++k) acc += dv[k]*w2[(size_t)k*DDIM];
  tb[(size_t)b*DDIM + n] = acc;
}

// ---------------- K9: W1 -> bf16, pre-packed into WMMA B-fragment order ----------------
// Fragment (kt, nt) covers K rows [kt*32, kt*32+32) x N cols [nt*16, nt*16+16).
// Lane L holds 16 contiguous bf16: element e = W1[kt*32 + (L/16)*16 + e][nt*16 + (L%16)].
// wp[((kt*64 + nt)*32 + L)*16 + e]
__global__ void k_w1pack(const float* __restrict__ updW, unsigned short* __restrict__ wp) {
  int i = blockIdx.x*256 + threadIdx.x;   // 0 .. 1M-1
  int e  = i & 15;
  int L  = (i >> 4) & 31;
  int nt = (i >> 9) & 63;
  int kt = i >> 15;
  int k = kt*32 + (L >> 4)*16 + e;
  int n = nt*16 + (L & 15);
  wp[i] = f2bf(updW[(size_t)k*DDIM + n]);
}

// ---------------- K10: fused WMMA GEMM + bias/ReLU + LayerNorm ----------------
// block: 256 thr = 8 waves; computes 16 rows x 1024 cols; wave w owns cols [w*128, w*128+128)
// A row-block (16x1024 bf16, 33KB LDS, padded rows) staged once; K loop is barrier-free.
__global__ void __launch_bounds__(256)
k_gemm_ln(const float* __restrict__ X, const unsigned short* __restrict__ wp,
          const float* __restrict__ tb, const float* __restrict__ gamma,
          const float* __restrict__ beta, float* __restrict__ out) {
  __shared__ unsigned short aAll[16][1032];        // row stride 516 dwords -> bank 4*lane
  __shared__ float sGamma[DDIM], sBeta[DDIM];
  __shared__ float redS[8][16], redQ[8][16];
  __shared__ float sMu[16], sRs[16];

  int r0 = blockIdx.x * 16;
  int tid  = threadIdx.x;
  int wave = tid >> 5, lane = tid & 31;
  int half = lane >> 4, lq = lane & 15;

  for (int t = tid; t < DDIM; t += 256) { sGamma[t] = gamma[t]; sBeta[t] = beta[t]; }

  // stage whole A row-block: 16 x 1024 f32 -> bf16 (float4 loads, 4x b16 stores)
  const float4* xv = reinterpret_cast<const float4*>(X + (size_t)r0*DDIM);
  for (int t = tid; t < 4096; t += 256) {
    int i = t >> 8, c4 = t & 255;
    float4 f = xv[(size_t)i*256 + c4];
    int k = c4*4;
    aAll[i][k]   = f2bf(f.x); aAll[i][k+1] = f2bf(f.y);
    aAll[i][k+2] = f2bf(f.z); aAll[i][k+3] = f2bf(f.w);
  }
  __syncthreads();

  v8f acc[8];
  #pragma unroll
  for (int j = 0; j < 8; ++j)
    #pragma unroll
    for (int v = 0; v < 8; ++v) acc[j][v] = 0.0f;

  const v16u* wpv = reinterpret_cast<const v16u*>(wp);
  for (int kt = 0; kt < 32; ++kt) {
    int kb = kt*32;
    // A fragment: two contiguous 8-element (16B) chunks per lane -> ds_load_b128 x2
    v16u au;
    #pragma unroll
    for (int e = 0; e < 8; ++e) au[e]   = aAll[lq][kb + half*16 + e];
    #pragma unroll
    for (int e = 0; e < 8; ++e) au[8+e] = aAll[lq][kb + half*16 + 8 + e];
    v16bf a = __builtin_bit_cast(v16bf, au);
    #pragma unroll
    for (int j = 0; j < 8; ++j) {                  // 8 N-tiles per wave
      v16u bu = wpv[(size_t)((kt*64 + wave*8 + j)*32 + lane)];  // 32B contiguous per lane
      v16bf bf = __builtin_bit_cast(v16bf, bu);
      acc[j] = __builtin_amdgcn_wmma_f32_16x16x32_bf16(
          false, a, false, bf, (short)0, acc[j], false, false);
    }
  }

  // epilogue: h = relu(acc + t[b][n]); accumulate sum / sumsq per row
  float s8[8], q8[8];
  #pragma unroll
  for (int v = 0; v < 8; ++v) { s8[v] = 0.f; q8[v] = 0.f; }
  #pragma unroll
  for (int j = 0; j < 8; ++j) {
    int n = wave*128 + j*16 + lq;
    #pragma unroll
    for (int v = 0; v < 8; ++v) {
      int r = r0 + v + half*8;
      int bb = r & (BDIM-1);                       // r = s*B + b
      float hv = fmaxf(acc[j][v] + tb[(size_t)bb*DDIM + n], 0.f);
      acc[j][v] = hv;
      s8[v] += hv; q8[v] += hv*hv;
    }
  }
  #pragma unroll
  for (int v = 0; v < 8; ++v) {                    // butterfly within 16-lane halves
    #pragma unroll
    for (int o = 1; o < 16; o <<= 1) {
      s8[v] += __shfl_xor(s8[v], o, 32);
      q8[v] += __shfl_xor(q8[v], o, 32);
    }
  }
  if (lq == 0) {
    #pragma unroll
    for (int v = 0; v < 8; ++v) {
      redS[wave][half*8 + v] = s8[v];
      redQ[wave][half*8 + v] = q8[v];
    }
  }
  __syncthreads();
  if (tid < 16) {
    float S = 0.f, Qq = 0.f;
    #pragma unroll
    for (int w = 0; w < 8; ++w) { S += redS[w][tid]; Qq += redQ[w][tid]; }
    float mu = S * (1.0f/DDIM);
    float var = Qq * (1.0f/DDIM) - mu*mu;
    sMu[tid] = mu;
    sRs[tid] = rsqrtf(var + 1e-5f);
  }
  __syncthreads();
  #pragma unroll
  for (int j = 0; j < 8; ++j) {
    int n = wave*128 + j*16 + lq;
    #pragma unroll
    for (int v = 0; v < 8; ++v) {
      int m = v + half*8;
      int r = r0 + m;
      out[(size_t)r*DDIM + n] = (acc[j][v] - sMu[m]) * sRs[m] * sGamma[n] + sBeta[n];
    }
  }
}

extern "C" void kernel_launch(void* const* d_in, const int* in_sizes, int n_in,
                              void* d_out, int out_size, void* d_ws, size_t ws_size,
                              hipStream_t stream) {
  (void)in_sizes; (void)n_in; (void)out_size; (void)ws_size;
  const float* input = (const float*)d_in[0];
  const float* gnf   = (const float*)d_in[1];
  const float* aWq   = (const float*)d_in[2];
  const float* abq   = (const float*)d_in[3];
  const float* aWk   = (const float*)d_in[4];
  const float* abk   = (const float*)d_in[5];
  const float* dWq   = (const float*)d_in[6];
  const float* dbq   = (const float*)d_in[7];
  const float* dWk   = (const float*)d_in[8];
  const float* dbk   = (const float*)d_in[9];
  const float* updW  = (const float*)d_in[10];
  const float* updb  = (const float*)d_in[11];
  const float* lng   = (const float*)d_in[12];
  const float* lnb   = (const float*)d_in[13];
  float* out = (float*)d_out;

  char* wsp = (char*)d_ws;
  auto alloc = [&](size_t bytes) {
    char* p = wsp; wsp += (bytes + 255) & ~(size_t)255; return p;
  };
  float* g      = (float*)alloc((size_t)BDIM*DDIM*4);
  float* common = (float*)alloc((size_t)BDIM*9*DDIM*4);
  float* Q      = (float*)alloc((size_t)BDIM*HDIM*DDIM*4);
  float* Qk     = (float*)alloc((size_t)BDIM*HDIM*DDIM*4);
  float* qc     = (float*)alloc((size_t)BDIM*HDIM*4);
  float* Mbuf   = (float*)alloc((size_t)BDIM*HDIM*NDIM*4);
  float* Qd     = (float*)alloc((size_t)BDIM*9*DDIM*4);
  float* Kd     = (float*)alloc((size_t)BDIM*9*DDIM*4);
  float* diffb  = (float*)alloc((size_t)BDIM*DDIM*4);
  float* tb     = (float*)alloc((size_t)BDIM*DDIM*4);
  unsigned short* w1 = (unsigned short*)alloc((size_t)DDIM*DDIM*2);

  k_mean   <<<BDIM*DDIM/256, 256, 0, stream>>>(input, g, common);
  k_qproj  <<<dim3(DDIM/64, BDIM/4, HDIM), 256, 0, stream>>>(g, aWq, abq, Q);
  k_qk     <<<dim3(DDIM/4, HDIM), 256, 0, stream>>>(Q, aWk, Qk);
  k_qc     <<<2, 256, 0, stream>>>(Q, abk, qc);
  k_logits <<<dim3(NDIM/32, BDIM), 256, 0, stream>>>(gnf, Qk, qc, Mbuf);
  k_softmax<<<BDIM*HDIM, 256, 0, stream>>>(Mbuf);
  k_closest<<<dim3(DDIM/32, BDIM), 256, 0, stream>>>(gnf, Mbuf, common);
  k_proj576<<<dim3(DDIM/64, (BDIM*9)/4), 256, 0, stream>>>(common, dWq, dbq, Qd);
  k_proj576<<<dim3(DDIM/64, (BDIM*9)/4), 256, 0, stream>>>(common, dWk, dbk, Kd);
  k_diffatt<<<BDIM, 256, 0, stream>>>(Qd, Kd, common, g, diffb);
  k_tcalc  <<<dim3(DDIM/256, BDIM), 256, 0, stream>>>(diffb, updW, updb, tb);
  k_w1pack <<<DDIM*DDIM/256, 256, 0, stream>>>(updW, w1);
  k_gemm_ln<<<MROWS/16, 256, 0, stream>>>(input, w1, tb, lng, lnb, out);
}
